// OptimizedAttention_13932873908509
// MI455X (gfx1250) — matrix operand order repaired and checked by reading, not compiled, not run on hardware
//
#include <hip/hip_runtime.h>

// ---------------------------------------------------------------------------
// CDNA5 (gfx1250) fused attention: bf16 WMMA GEMMs + flash attention.
// Workspace layout (bf16 elements), total ~76 MiB (fits comfortably in L2):
//   Xb    [4096][2048]          hidden in bf16
//   Wqkvt [3072][2048]          [Wq^T ; Wk^T ; Wv^T], row = output col
//   Wot   [2048][2048]          Wo^T
//   Qr    [B][NH][S][DH]        rope'd in place
//   Kr    [B][NKV][S][DH]       rope'd in place
//   Vt    [B][NKV][DH][S]       transposed for contiguous P*V B-fragments
//   Ao    [4096][2048]          attention output (bf16)
// ---------------------------------------------------------------------------

typedef __attribute__((ext_vector_type(16))) __bf16    v16bf;
typedef __attribute__((ext_vector_type(8)))  float     v8f;
typedef __attribute__((ext_vector_type(4)))  unsigned  u32x4;   // POD 16B vector

union Frag16 {
    v16bf    v;
    unsigned u[8];
    u32x4    q[2];
};

__device__ __forceinline__ v8f wmma_bf16(v16bf a, v16bf b, v8f c) {
    // D = A(16x32 bf16) * B(32x16 bf16) + C(16x16 f32)
    return __builtin_amdgcn_wmma_f32_16x16x32_bf16(
        false, a, false, b, (short)0, c, false, false);
}

// A-fragment (16x32, row-major source, one row per lane16).
// Contiguous k order used here (A and B match): lanes 0-15 M=0..15; dword d holds K pair
//   k = 2d + 16*half   (half = lane/16)
// Dword indices are contiguous (0..7 for half=0) -> 2x ds_load_b128.
__device__ __forceinline__ v16bf load_a_pairs(const __bf16* rowPtr, int half, int kbase) {
    Frag16 f;
    const unsigned* p = (const unsigned*)rowPtr;
#pragma unroll
    for (int d = 0; d < 8; ++d) {
        int kidx = (kbase >> 1) + d + 8 * half;
        f.u[d] = p[kidx];
    }
    return f.v;
}

// B-fragment (32x16) from a K-contiguous row (source stored [n][k]).
// Contiguous k order used here (matches A): lane n holds col n%16; element e -> k = 16*(lane/16) + e,
// i.e. 16 contiguous bf16 = 32 contiguous bytes per lane.
__device__ __forceinline__ v16bf load_b_contig(const __bf16* rowPtr, int half, int kbase) {
    Frag16 f;
    const u32x4* p = (const u32x4*)(rowPtr + kbase + 16 * half);
    f.q[0] = p[0];
    f.q[1] = p[1];
    return f.v;
}

// ------------------------------ conversions --------------------------------

__global__ void cvt_x_kernel(const float* __restrict__ X, __bf16* __restrict__ Xb, int n) {
    int i = (blockIdx.x * blockDim.x + threadIdx.x) * 2;
    if (i < n) {
        Xb[i]     = (__bf16)X[i];
        Xb[i + 1] = (__bf16)X[i + 1];
    }
}

// W [K][Nw] f32 (row-major)  ->  Wt [Nw][K] bf16
__global__ void cvt_wt_kernel(const float* __restrict__ W, __bf16* __restrict__ Wt,
                              int K, int Nw) {
    int i = blockIdx.x * blockDim.x + threadIdx.x;
    if (i >= K * Nw) return;
    int k = i % K;
    int n = i / K;
    Wt[(size_t)n * K + k] = (__bf16)W[(size_t)k * Nw + n];
}

// ------------------------------ GEMM (bf16) --------------------------------
// Block tile 128(M) x 128(N), K step 32, double-buffered LDS (one barrier per
// step). 8 waves arranged 4(M) x 2(N); each wave computes 32x64 = 8 WMMAs per
// K-step from 2 A-fragments and 4 B-fragments.
// mode 0: plain f32 store to outF (M x N row-major)
// mode 1: QKV scatter epilogue (Q/K row-major per head, V transposed)

__global__ __launch_bounds__(256) void gemm_bf16_kernel(
    const __bf16* __restrict__ A, const __bf16* __restrict__ Bt,
    int M, int N, int K, int mode,
    float* __restrict__ outF,
    __bf16* __restrict__ Qr, __bf16* __restrict__ Kr, __bf16* __restrict__ Vt)
{
    __shared__ __bf16 As[2][128][32];
    __shared__ __bf16 Bs[2][128][32];

    const int t = threadIdx.x;
    const int lane = t & 31, w = t >> 5;
    const int half = lane >> 4, lane16 = lane & 15;
    const int wm = w & 3, wn = w >> 2;
    const int blockM = blockIdx.y * 128;
    const int blockN = blockIdx.x * 128;

    v8f acc[2][4];
#pragma unroll
    for (int i = 0; i < 2; ++i)
#pragma unroll
        for (int j = 0; j < 4; ++j)
#pragma unroll
            for (int e = 0; e < 8; ++e) acc[i][j][e] = 0.0f;

    // Tile loaders: 128 rows x 32 cols each; thread t -> row t/2, 16 elems.
    const int ldrow = t >> 1, ldcol = (t & 1) * 16;
    const __bf16* gA = A  + (size_t)(blockM + ldrow) * K + ldcol;
    const __bf16* gB = Bt + (size_t)(blockN + ldrow) * K + ldcol;

    // prologue: stage tile 0
    u32x4 ra0 = ((const u32x4*)gA)[0], ra1 = ((const u32x4*)gA)[1];
    u32x4 rb0 = ((const u32x4*)gB)[0], rb1 = ((const u32x4*)gB)[1];
    *(u32x4*)&As[0][ldrow][ldcol]     = ra0;
    *(u32x4*)&As[0][ldrow][ldcol + 8] = ra1;
    *(u32x4*)&Bs[0][ldrow][ldcol]     = rb0;
    *(u32x4*)&Bs[0][ldrow][ldcol + 8] = rb1;
    __syncthreads();

    const int steps = K >> 5;
    for (int i = 0; i < steps; ++i) {
        const int cur = i & 1, nxt = cur ^ 1;

        // issue next tile's global loads (overlap with WMMA below)
        if (i + 1 < steps) {
            const __bf16* pa = gA + (size_t)(i + 1) * 32;
            const __bf16* pb = gB + (size_t)(i + 1) * 32;
            ra0 = ((const u32x4*)pa)[0]; ra1 = ((const u32x4*)pa)[1];
            rb0 = ((const u32x4*)pb)[0]; rb1 = ((const u32x4*)pb)[1];
        }

        v16bf a0 = load_a_pairs(&As[cur][wm * 32 + lane16][0],      half, 0);
        v16bf a1 = load_a_pairs(&As[cur][wm * 32 + 16 + lane16][0], half, 0);
#pragma unroll
        for (int c = 0; c < 4; ++c) {
            v16bf bf = load_b_contig(&Bs[cur][wn * 64 + c * 16 + lane16][0], half, 0);
            acc[0][c] = wmma_bf16(a0, bf, acc[0][c]);
            acc[1][c] = wmma_bf16(a1, bf, acc[1][c]);
        }

        if (i + 1 < steps) {
            *(u32x4*)&As[nxt][ldrow][ldcol]     = ra0;
            *(u32x4*)&As[nxt][ldrow][ldcol + 8] = ra1;
            *(u32x4*)&Bs[nxt][ldrow][ldcol]     = rb0;
            *(u32x4*)&Bs[nxt][ldrow][ldcol + 8] = rb1;
        }
        __syncthreads();
    }

#pragma unroll
    for (int cm = 0; cm < 2; ++cm)
#pragma unroll
        for (int cn = 0; cn < 4; ++cn)
#pragma unroll
            for (int r = 0; r < 8; ++r) {
                int m = blockM + wm * 32 + cm * 16 + r + 8 * half;
                int n = blockN + wn * 64 + cn * 16 + lane16;
                float vl = acc[cm][cn][r];
                if (mode == 0) {
                    outF[(size_t)m * N + n] = vl;
                } else {
                    int b = m >> 11, s = m & 2047;      // S = 2048
                    if (n < 2048) {                      // Q: head = n/64
                        int h = n >> 6, d = n & 63;
                        Qr[(((size_t)(b * 32 + h) * 2048) + s) * 64 + d] = (__bf16)vl;
                    } else if (n < 2560) {               // K
                        int kvh = (n - 2048) >> 6, d = n & 63;
                        Kr[(((size_t)(b * 8 + kvh) * 2048) + s) * 64 + d] = (__bf16)vl;
                    } else {                             // V, transposed [DH][S]
                        int kvh = (n - 2560) >> 6, d = n & 63;
                        Vt[(((size_t)(b * 8 + kvh) * 64) + d) * 2048 + s] = (__bf16)vl;
                    }
                }
            }
}

// ------------------------------- RoPE --------------------------------------
// In-place on Qr and Kr: x'[j] = x[j]*cos - x[j+32]*sin ; x'[j+32] = x[j+32]*cos + x[j]*sin

__global__ void rope_kernel(const int* __restrict__ pos_ids,
                            __bf16* __restrict__ Qr, __bf16* __restrict__ Kr) {
    int idx = blockIdx.x * blockDim.x + threadIdx.x;
    if (idx >= 2 * 2048 * 40 * 32) return;       // B*S*(NH+NKV)*(DH/2)
    int j  = idx & 31;
    int hs = (idx >> 5) % 40;
    int sp = (idx >> 5) / 40;                    // b*S + s
    int b = sp >> 11, s = sp & 2047;
    float pos = (float)pos_ids[sp];
    // inv_freq = theta^(-2j/64) = exp(-(2j/64)*ln(10000))
    float inv = __expf(-(float)(2 * j) * (9.210340371976184f / 64.0f));
    float sn, cs;
    __sincosf(pos * inv, &sn, &cs);
    __bf16* ptr;
    if (hs < 32) ptr = Qr + (((size_t)(b * 32 + hs) * 2048) + s) * 64;
    else         ptr = Kr + (((size_t)(b * 8 + (hs - 32)) * 2048) + s) * 64;
    float x1 = (float)ptr[j], x2 = (float)ptr[j + 32];
    ptr[j]      = (__bf16)(x1 * cs - x2 * sn);
    ptr[j + 32] = (__bf16)(x2 * cs + x1 * sn);
}

// --------------------------- flash attention -------------------------------
// One wave per 16-query tile; 32 keys per iteration; online softmax.
// C-layout: row = r + 8*half, col = lane16 -> row reductions are 16-lane shuffles.
// Waves are fully independent (different causal trip counts -> no barriers).

__global__ __launch_bounds__(256) void flash_attn_kernel(
    const __bf16* __restrict__ Qr, const __bf16* __restrict__ Kr,
    const __bf16* __restrict__ Vt, __bf16* __restrict__ Ao)
{
    __shared__ __bf16 pbuf[8][16 * 32];          // per-wave P tile (16x32 bf16)

    const int t = threadIdx.x, lane = t & 31, w = t >> 5;
    const int half = lane >> 4, lane16 = lane & 15;
    const int q0 = blockIdx.x * 128 + w * 16;
    const int h = blockIdx.y, b = blockIdx.z;
    const int kvh = h >> 2;                      // N_REP = 4

    const __bf16* Qbase = Qr + (((size_t)(b * 32 + h) * 2048) + q0) * 64;
    const __bf16* Kbase = Kr + ((size_t)(b * 8 + kvh) * 2048) * 64;
    const __bf16* Vbase = Vt + ((size_t)(b * 8 + kvh) * 64) * 2048;

    // Q fragments for this wave's 16 rows (DH=64 -> two K-blocks of 32)
    const __bf16* qrow = Qbase + (size_t)lane16 * 64;
    v16bf qf0 = load_a_pairs(qrow, half, 0);
    v16bf qf1 = load_a_pairs(qrow, half, 32);

    v8f out[4];
    float mrow[8], lrow[8];
#pragma unroll
    for (int c = 0; c < 4; ++c)
#pragma unroll
        for (int e = 0; e < 8; ++e) out[c][e] = 0.0f;
#pragma unroll
    for (int r = 0; r < 8; ++r) { mrow[r] = -3.0e38f; lrow[r] = 0.0f; }

    const int nkv = q0 + 16;                     // causal extent for this tile
    for (int kb = 0; kb < nkv; kb += 32) {
        // ---- scores: Q(16x64) @ K^T(64x32) -> two 16x16 f32 tiles ----
        v8f s0, s1;
#pragma unroll
        for (int e = 0; e < 8; ++e) { s0[e] = 0.0f; s1[e] = 0.0f; }
        const __bf16* krow0 = Kbase + (size_t)(kb + lane16) * 64;
        const __bf16* krow1 = Kbase + (size_t)(kb + 16 + lane16) * 64;
        v16bf kf00 = load_b_contig(krow0, half, 0);
        v16bf kf01 = load_b_contig(krow0, half, 32);
        v16bf kf10 = load_b_contig(krow1, half, 0);
        v16bf kf11 = load_b_contig(krow1, half, 32);
        s0 = wmma_bf16(qf0, kf00, s0);
        s0 = wmma_bf16(qf1, kf01, s0);
        s1 = wmma_bf16(qf0, kf10, s1);
        s1 = wmma_bf16(qf1, kf11, s1);

        if (kb + 32 < nkv)                       // gfx1250 global_prefetch_b8
            __builtin_prefetch(Kbase + (size_t)(kb + 32 + lane) * 64, 0, 1);

        // ---- scale + causal mask + online softmax ----
        const float scale = 0.125f;              // 1/sqrt(64)
        int key0 = kb + lane16, key1 = kb + 16 + lane16;
#pragma unroll
        for (int r = 0; r < 8; ++r) {
            int qidx = q0 + r + 8 * half;
            float x0 = (key0 > qidx) ? -1e30f : s0[r] * scale;
            float x1 = (key1 > qidx) ? -1e30f : s1[r] * scale;
            float tm = fmaxf(x0, x1);
#pragma unroll
            for (int mk = 1; mk < 16; mk <<= 1) tm = fmaxf(tm, __shfl_xor(tm, mk, 32));
            float mnew  = fmaxf(mrow[r], tm);
            float alpha = __expf(mrow[r] - mnew);
            mrow[r] = mnew;
            float p0 = __expf(x0 - mnew);
            float p1 = __expf(x1 - mnew);
            float ps = p0 + p1;
#pragma unroll
            for (int mk = 1; mk < 16; mk <<= 1) ps += __shfl_xor(ps, mk, 32);
            lrow[r] = lrow[r] * alpha + ps;
#pragma unroll
            for (int c = 0; c < 4; ++c) out[c][r] *= alpha;
            // stage P (C-layout -> row-major LDS) for the A-fragment reload
            pbuf[w][(r + 8 * half) * 32 + lane16]      = (__bf16)p0;
            pbuf[w][(r + 8 * half) * 32 + 16 + lane16] = (__bf16)p1;
        }
        asm volatile("" ::: "memory");           // order LDS store -> load (same wave, DS in-order)

        // ---- P(16x32) @ V(32x64): A from LDS, B contiguous from Vt ----
        v16bf pf = load_a_pairs(&pbuf[w][lane16 * 32], half, 0);
#pragma unroll
        for (int c = 0; c < 4; ++c) {
            const __bf16* vrow = Vbase + (size_t)(c * 16 + lane16) * 2048 + kb;
            v16bf vf = load_b_contig(vrow, half, 0);
            out[c] = wmma_bf16(pf, vf, out[c]);
        }
    }

    // ---- epilogue: normalize and store bf16 to Ao[b][q][h*64+d] ----
#pragma unroll
    for (int c = 0; c < 4; ++c)
#pragma unroll
        for (int r = 0; r < 8; ++r) {
            int q = q0 + r + 8 * half;
            int d = c * 16 + lane16;
            Ao[(((size_t)b * 2048 + q) * 2048) + h * 64 + d] =
                (__bf16)(out[c][r] / lrow[r]);
        }
}

// ------------------------------ launcher -----------------------------------

extern "C" void kernel_launch(void* const* d_in, const int* in_sizes, int n_in,
                              void* d_out, int out_size, void* d_ws, size_t ws_size,
                              hipStream_t stream) {
    const float* hidden = (const float*)d_in[0];
    const int*   pos    = (const int*)d_in[1];
    const float* Wq     = (const float*)d_in[2];
    const float* Wk     = (const float*)d_in[3];
    const float* Wv     = (const float*)d_in[4];
    const float* Wo     = (const float*)d_in[5];
    float*       out    = (float*)d_out;

    // workspace carve-up (bf16 elements); ~76 MiB total
    __bf16* p     = (__bf16*)d_ws;
    __bf16* Xb    = p; p += (size_t)4096 * 2048;
    __bf16* Wqkvt = p; p += (size_t)3072 * 2048;
    __bf16* Wot   = p; p += (size_t)2048 * 2048;
    __bf16* Qr    = p; p += (size_t)2 * 32 * 2048 * 64;
    __bf16* Kr    = p; p += (size_t)2 * 8 * 2048 * 64;
    __bf16* Vt    = p; p += (size_t)2 * 8 * 64 * 2048;
    __bf16* Ao    = p; p += (size_t)4096 * 2048;

    // 1) conversions
    {
        int n = 4096 * 2048;
        cvt_x_kernel<<<(n / 2 + 255) / 256, 256, 0, stream>>>(hidden, Xb, n);
    }
    cvt_wt_kernel<<<(2048 * 2048 + 255) / 256, 256, 0, stream>>>(Wq, Wqkvt,                       2048, 2048);
    cvt_wt_kernel<<<(2048 * 512  + 255) / 256, 256, 0, stream>>>(Wk, Wqkvt + (size_t)2048 * 2048, 2048, 512);
    cvt_wt_kernel<<<(2048 * 512  + 255) / 256, 256, 0, stream>>>(Wv, Wqkvt + (size_t)2560 * 2048, 2048, 512);
    cvt_wt_kernel<<<(2048 * 2048 + 255) / 256, 256, 0, stream>>>(Wo, Wot,                         2048, 2048);

    // 2) fused QKV GEMM with scatter epilogue (M=4096, N=3072, K=2048)
    gemm_bf16_kernel<<<dim3(3072 / 128, 4096 / 128), 256, 0, stream>>>(
        Xb, Wqkvt, 4096, 3072, 2048, /*mode=*/1, nullptr, Qr, Kr, Vt);

    // 3) RoPE in place on Q and K
    {
        int n = 2 * 2048 * 40 * 32;
        rope_kernel<<<(n + 255) / 256, 256, 0, stream>>>(pos, Qr, Kr);
    }

    // 4) flash attention -> Ao
    flash_attn_kernel<<<dim3(2048 / 128, 32, 2), 256, 0, stream>>>(Qr, Kr, Vt, Ao);

    // 5) output projection (M=4096, N=2048, K=2048), f32 result
    gemm_bf16_kernel<<<dim3(2048 / 128, 4096 / 128), 256, 0, stream>>>(
        Ao, Wot, 4096, 2048, 2048, /*mode=*/0, out, nullptr, nullptr, nullptr);
}
